// Encoder_57990648430636
// MI455X (gfx1250) — matrix operand order repaired and checked by reading, compiled, not run on hardware
//
#include <hip/hip_runtime.h>
#include <hip/hip_bf16.h>
#include <stdint.h>

#define D      256
#define KSTEP  32
#define NKS    (D / KSTEP)   // 8 K-steps
#define MTILE  64
#define BLOCK  256

typedef __attribute__((ext_vector_type(16))) __bf16 v16bf;
typedef __attribute__((ext_vector_type(8)))  float  v8f;

__device__ __forceinline__ unsigned f32_to_bf16_bits(float f) {
    unsigned u = __float_as_uint(f);
    return (u + 0x7FFFu + ((u >> 16) & 1u)) >> 16;   // RNE
}
__device__ __forceinline__ unsigned pack_bf16x2(float lo, float hi) {
    return f32_to_bf16_bits(lo) | (f32_to_bf16_bits(hi) << 16);
}
__device__ __forceinline__ float bf16_bits_to_f32(unsigned short h) {
    return __uint_as_float(((unsigned)h) << 16);
}

// ---------------------------------------------------------------------------
// Pack a 256x256 fp32 weight matrix (row = K, col = N) into bf16 WMMA
// B-fragments: out[((ks*16+nt)*512) + lane*16 + e]; each lane's 16-element
// fragment is one contiguous 32-byte chunk.
// B layout (16-bit, 32x16): lanes 0-15 hold K=0..7,16..23 (2 per VGPR), lanes 16-31
// hold K=8..15,24..31  =>  k_local = 8*(lane>>4) + (e&7) + 16*(e>>3).
// ---------------------------------------------------------------------------
__global__ void pack_weight_kernel(const float* __restrict__ W,
                                   unsigned short* __restrict__ out) {
    int idx  = blockIdx.x * blockDim.x + threadIdx.x;  // 0..65535
    int e    = idx & 15;
    int lane = (idx >> 4) & 31;
    int frag = idx >> 9;           // 0..127
    int nt   = frag & 15;          // n-tile
    int ks   = frag >> 4;          // k-step
    int n    = nt * 16 + (lane & 15);
    int kl   = ((lane >> 4) << 3) + (e & 7) + ((e >> 3) << 4);
    int k    = ks * KSTEP + kl;
    out[idx] = (unsigned short)f32_to_bf16_bits(W[k * D + n]);
}

__global__ void zero_f32_kernel(float* __restrict__ p, int n) {
    int i = blockIdx.x * blockDim.x + threadIdx.x;
    if (i < n) p[i] = 0.0f;
}

// ---------------------------------------------------------------------------
// GEMM: out = relu(A1 @ B1 [+ A2 @ B2] + bias),  K = N = 256.
// A* fp32 row-major [M,256] in global -> bf16 A-fragments staged via LDS.
// B* pre-packed bf16 fragments.  Block = 256 thr = 8 waves computes a 64x256
// tile: wave w -> m-subtile (w&3), n-half (w>>2), 8 accumulators.
//
// A-frag layout (16-bit, 16x32): lanes0-15: V0-3=K0..7,  V4-7=K16..23;
//                                lanes16-31: V0-3=K8..15, V4-7=K24..31.
// Inverse (kl -> lane-half/elem): lh=(kl>>3)&1, e=(kl&7)|((kl&16)>>1).
// Per thread: one row (ml=tid>>2), 8 consecutive K (kl=(tid&3)*8..+7) which
// map to 8 consecutive LDS elements -> 2x b128 load, 1x ds_store_b128.
// Rows >= M are clamped (not zeroed): A row m only feeds D row m, and those
// rows are never stored, so no exec-mask guard is needed on the load path.
// ---------------------------------------------------------------------------
template <bool DUAL, bool OUT_BF16>
__global__ void __launch_bounds__(BLOCK)
gemm_wmma_kernel(const float* __restrict__ A1,
                 const unsigned short* __restrict__ B1,
                 const float* __restrict__ A2,
                 const unsigned short* __restrict__ B2,
                 const float* __restrict__ bias,
                 float* __restrict__ outF,
                 unsigned short* __restrict__ outB,
                 int M) {
    __shared__ __align__(32) unsigned short ldsA[(DUAL ? 2 : 1)][4][32][16];

    const int tid  = threadIdx.x;
    const int lane = tid & 31;
    const int w    = tid >> 5;      // 0..7
    const int mt   = w & 3;         // m-subtile (16 rows)
    const int nh   = w >> 2;        // n-half (128 cols)
    const int m0   = blockIdx.x * MTILE;

    // Per-thread constants for the A-pack path.
    const int ml  = tid >> 2;                  // tile row 0..63
    const int q   = tid & 3;                   // k-quarter
    const int kq  = q * 8;                     // kl base (0,8,16,24)
    const int lh  = q & 1;
    const int e0  = (q >> 1) * 8;
    const int dl  = (ml & 15) | (lh << 4);
    const int smt = ml >> 4;
    int arow = m0 + ml;
    if (arow >= M) arow = M - 1;               // clamp: padded rows unused
    const float4* __restrict__ pA1 = (const float4*)(A1 + (size_t)arow * D);
    const float4* __restrict__ pA2 =
        DUAL ? (const float4*)(A2 + (size_t)arow * D) : nullptr;

    v8f acc[8];
#pragma unroll
    for (int j = 0; j < 8; ++j)
#pragma unroll
        for (int r = 0; r < 8; ++r) acc[j][r] = 0.0f;

    for (int ks = 0; ks < NKS; ++ks) {
        const int f4 = (ks * KSTEP + kq) >> 2;   // float4 index into the row
        {
            float4 a = pA1[f4];
            float4 b = pA1[f4 + 1];
            uint4 p;
            p.x = pack_bf16x2(a.x, a.y);
            p.y = pack_bf16x2(a.z, a.w);
            p.z = pack_bf16x2(b.x, b.y);
            p.w = pack_bf16x2(b.z, b.w);
            *(uint4*)&ldsA[0][smt][dl][e0] = p;
        }
        if (DUAL) {
            float4 a = pA2[f4];
            float4 b = pA2[f4 + 1];
            uint4 p;
            p.x = pack_bf16x2(a.x, a.y);
            p.y = pack_bf16x2(a.z, a.w);
            p.z = pack_bf16x2(b.x, b.y);
            p.w = pack_bf16x2(b.z, b.w);
            *(uint4*)&ldsA[1][smt][dl][e0] = p;
        }
        __syncthreads();

        v16bf a1 = *(const v16bf*)&ldsA[0][mt][lane][0];
        v16bf a2;
        if (DUAL) a2 = *(const v16bf*)&ldsA[1][mt][lane][0];

#pragma unroll
        for (int j = 0; j < 8; ++j) {
            int nt = nh * 8 + j;
            const v16bf b1 =
                *(const v16bf*)&B1[((size_t)(ks * 16 + nt)) * 512 + lane * 16];
            acc[j] = __builtin_amdgcn_wmma_f32_16x16x32_bf16(
                false, a1, false, b1, (short)0, acc[j], false, false);
            if (DUAL) {
                const v16bf b2 =
                    *(const v16bf*)&B2[((size_t)(ks * 16 + nt)) * 512 + lane * 16];
                acc[j] = __builtin_amdgcn_wmma_f32_16x16x32_bf16(
                    false, a2, false, b2, (short)0, acc[j], false, false);
            }
        }
        __syncthreads();
    }

    // Epilogue: D layout f32 16x16 -> VGPR r: m = r + 8*(lane>=16), n = lane&15
    const int rbase = m0 + mt * 16 + ((lane >> 4) << 3);
    const int ncol  = lane & 15;
#pragma unroll
    for (int j = 0; j < 8; ++j) {
        int n = (nh * 8 + j) * 16 + ncol;
        float bv = bias[n];
#pragma unroll
        for (int r = 0; r < 8; ++r) {
            int row = rbase + r;
            if (row < M) {
                float v = acc[j][r] + bv;
                v = v > 0.0f ? v : 0.0f;
                if (OUT_BF16)
                    outB[(size_t)row * D + n] = (unsigned short)f32_to_bf16_bits(v);
                else
                    outF[(size_t)row * D + n] = v;
            }
        }
    }
}

// ---------------------------------------------------------------------------
// Edge scatter-max: msg = h[src]*w (all >= 0 since h=relu(..), w in [0,1)),
// so u32 atomicMax on raw float bits == float max.  agg pre-zeroed (== the
// reference's where(isfinite, segment_max, 0) since messages are >= 0).
// One wave per edge; each lane handles 8 contiguous features (1x b128 load).
// ---------------------------------------------------------------------------
__global__ void scatter_max_kernel(const unsigned short* __restrict__ h,
                                   const int* __restrict__ src,
                                   const int* __restrict__ dst,
                                   const float* __restrict__ w,
                                   unsigned int* __restrict__ agg,
                                   int E) {
    int edge = blockIdx.x * 8 + (threadIdx.x >> 5);
    if (edge >= E) return;
    int lane = threadIdx.x & 31;
    int s = src[edge];
    int d = dst[edge];
    float wt = w[edge];
    const uint4 raw = *(const uint4*)(h + (size_t)s * D + lane * 8);
    unsigned int* ap = agg + (size_t)d * D + lane * 8;
    unsigned vals[4] = {raw.x, raw.y, raw.z, raw.w};
#pragma unroll
    for (int q = 0; q < 4; ++q) {
        float lo = bf16_bits_to_f32((unsigned short)(vals[q] & 0xFFFFu)) * wt;
        float hi = bf16_bits_to_f32((unsigned short)(vals[q] >> 16)) * wt;
        atomicMax(&ap[q * 2 + 0], __float_as_uint(lo));
        atomicMax(&ap[q * 2 + 1], __float_as_uint(hi));
    }
}

extern "C" void kernel_launch(void* const* d_in, const int* in_sizes, int n_in,
                              void* d_out, int out_size, void* d_ws, size_t ws_size,
                              hipStream_t stream) {
    const float* x    = (const float*)d_in[0];
    const int*   src0 = (const int*)d_in[1];
    const int*   dst0 = (const int*)d_in[2];
    const float* w0   = (const float*)d_in[3];
    const int*   src1 = (const int*)d_in[4];
    const int*   dst1 = (const int*)d_in[5];
    const float* w1   = (const float*)d_in[6];
    const float* Wmat[6] = {(const float*)d_in[7],   // Wp1
                            (const float*)d_in[9],   // Ws1
                            (const float*)d_in[10],  // Wn1
                            (const float*)d_in[12],  // Wp2
                            (const float*)d_in[14],  // Ws2
                            (const float*)d_in[15]}; // Wn2
    const float* bp1 = (const float*)d_in[8];
    const float* b1  = (const float*)d_in[11];
    const float* bp2 = (const float*)d_in[13];
    const float* b2  = (const float*)d_in[16];

    const int N0 = 100000, N1 = 20000, N2 = 4000;
    const int E0 = in_sizes[1], E1 = in_sizes[4];

    // workspace carve-up (256B aligned)
    char* ws = (char*)d_ws;
    size_t off = 0;
    auto carve = [&](size_t bytes) -> void* {
        void* p = ws + off;
        off = (off + bytes + 255) & ~(size_t)255;
        return p;
    };
    unsigned short* wpack[6];
    for (int i = 0; i < 6; ++i)
        wpack[i] = (unsigned short*)carve((size_t)D * D * sizeof(unsigned short));
    unsigned short* h1   = (unsigned short*)carve((size_t)N0 * D * 2);  // bf16
    float*          agg1 = (float*)carve((size_t)N1 * D * 4);
    float*          out1 = (float*)carve((size_t)N1 * D * 4);
    unsigned short* h2   = (unsigned short*)carve((size_t)N1 * D * 2);  // bf16
    float*          agg2 = (float*)carve((size_t)N2 * D * 4);

    // 1) pack weights fp32 -> bf16 WMMA fragments
    for (int i = 0; i < 6; ++i)
        pack_weight_kernel<<<(D * D) / BLOCK, BLOCK, 0, stream>>>(Wmat[i], wpack[i]);

    // 2) zero the segment-max accumulators
    zero_f32_kernel<<<(N1 * D) / BLOCK, BLOCK, 0, stream>>>(agg1, N1 * D);
    zero_f32_kernel<<<(N2 * D) / BLOCK, BLOCK, 0, stream>>>(agg2, N2 * D);

    // 3) h1 = relu(x @ Wp1 + bp1)  (bf16 out, only feeds the gather)
    gemm_wmma_kernel<false, true><<<(N0 + MTILE - 1) / MTILE, BLOCK, 0, stream>>>(
        x, wpack[0], nullptr, nullptr, bp1, nullptr, h1, N0);

    // 4) agg1 = segment_max(h1[src0] * w0, dst0)
    scatter_max_kernel<<<(E0 + 7) / 8, BLOCK, 0, stream>>>(
        h1, src0, dst0, w0, (unsigned int*)agg1, E0);

    // 5) out1 = relu(x[:N1] @ Ws1 + agg1 @ Wn1 + b1)
    gemm_wmma_kernel<true, false><<<(N1 + MTILE - 1) / MTILE, BLOCK, 0, stream>>>(
        x, wpack[1], agg1, wpack[2], b1, out1, nullptr, N1);

    // 6) h2 = relu(out1 @ Wp2 + bp2)  (bf16 out, only feeds the gather)
    gemm_wmma_kernel<false, true><<<(N1 + MTILE - 1) / MTILE, BLOCK, 0, stream>>>(
        out1, wpack[3], nullptr, nullptr, bp2, nullptr, h2, N1);

    // 7) agg2 = segment_max(h2[src1] * w1, dst1)
    scatter_max_kernel<<<(E1 + 7) / 8, BLOCK, 0, stream>>>(
        h2, src1, dst1, w1, (unsigned int*)agg2, E1);

    // 8) d_out = relu(out1[:N2] @ Ws2 + agg2 @ Wn2 + b2)
    gemm_wmma_kernel<true, false><<<(N2 + MTILE - 1) / MTILE, BLOCK, 0, stream>>>(
        out1, wpack[4], agg2, wpack[5], b2, (float*)d_out, nullptr, N2);

    (void)n_in; (void)out_size; (void)ws_size;
}
